// MultiHeadAttentionBlock_29403346109024
// MI455X (gfx1250) — matrix operand order repaired and checked by reading, not compiled, not run on hardware
//
#include <hip/hip_runtime.h>

#define DMODEL 1024
#define NHEAD  16
#define DK     64
#define SEQ    2048
#define BATCH  4

typedef __attribute__((ext_vector_type(16))) __bf16 bf16x16;
typedef __attribute__((ext_vector_type(8)))  __bf16 bf16x8;
typedef __attribute__((ext_vector_type(4)))  __bf16 bf16x4;
typedef __attribute__((ext_vector_type(8)))  float  f32x8;
typedef __attribute__((ext_vector_type(4)))  float  f32x4;

// gcc-style vector: matches the builtin's parameter type exactly
typedef int i32v4 __attribute__((vector_size(16)));

union ABf { bf16x16 v; bf16x8 h[2]; };

#define LDS_AS __attribute__((address_space(3)))

#if defined(__has_builtin)
#  if __has_builtin(__builtin_amdgcn_global_load_async_to_lds_b128)
#    define HAVE_ASYNC_LDS 1
#  endif
#endif
#ifndef HAVE_ASYNC_LDS
#  define HAVE_ASYNC_LDS 0
#endif

__device__ __forceinline__ void async_copy_b128(const void* gsrc, void* ldst) {
#if HAVE_ASYNC_LDS
    __builtin_amdgcn_global_load_async_to_lds_b128(
        (i32v4*)(void*)gsrc, (LDS_AS i32v4*)(void*)ldst, 0, 0);
#else
    *(bf16x8*)ldst = *(const bf16x8*)gsrc;
#endif
}

__device__ __forceinline__ void async_wait0() {
#if HAVE_ASYNC_LDS
#  if defined(__has_builtin) && __has_builtin(__builtin_amdgcn_s_wait_asynccnt)
    __builtin_amdgcn_s_wait_asynccnt(0);
#  else
    asm volatile("s_wait_asynccnt 0" ::: "memory");
#  endif
#endif
}

__device__ __forceinline__ f32x8 wmma_bf16(bf16x16 a, bf16x16 b, f32x8 c) {
    // 8 args: (neg_a, A, neg_b, B, c_mod, C, reuse_a, reuse_b)
    return __builtin_amdgcn_wmma_f32_16x16x32_bf16(false, a, false, b, (short)0, c,
                                                   false, false);
}

// ---------------------------------------------------------------------------
// f32 -> bf16 elementwise cast (weights only)
// ---------------------------------------------------------------------------
__global__ __launch_bounds__(256) void cast_f32_bf16(const float* __restrict__ in,
                                                     __bf16* __restrict__ out, int n) {
    int i = (blockIdx.x * blockDim.x + threadIdx.x) * 4;
    if (i + 3 < n) {
        f32x4 v = *(const f32x4*)(in + i);
        bf16x4 o;
        o[0] = (__bf16)v.x; o[1] = (__bf16)v.y; o[2] = (__bf16)v.z; o[3] = (__bf16)v.w;
        *(bf16x4*)(out + i) = o;
    }
}

// ---------------------------------------------------------------------------
// GEMM: C[M,N] = A[M,K](f32, cast->bf16 in staging) * B[N,K](bf16)^T
// MODE 0: bf16 head-split out[b,h,s,d]  |  MODE 1: f32 row-major out[m,n]
// Block tile 128x128x64, 8 wave32 waves, wave tile 64x32 (4x2 WMMA accums).
// Pipelined: next A tile prefetched to registers, next B tile async-copied
// into double-buffered LDS while current tile is consumed by WMMA.
// ---------------------------------------------------------------------------
template <int MODE>
__global__ __launch_bounds__(256) void gemm_bf16_wmma(const float* __restrict__ A,
                                                      const __bf16* __restrict__ B,
                                                      void* __restrict__ out,
                                                      int M, int N, int K) {
    constexpr int BM = 128, BN = 128, BK = 64, LDT = BK + 8;
    __shared__ __bf16 As[BM * LDT];          // single buffer (needs f32->bf16)
    __shared__ __bf16 Bs[2][BN * LDT];       // double buffer, async-filled

    const int tid   = threadIdx.x;
    const int wave  = tid >> 5;
    const int lane  = tid & 31;
    const int l16   = lane & 15;
    const int lhalf = lane >> 4;
    const int wm    = wave >> 2;   // 0..1
    const int wn    = wave & 3;    // 0..3
    const int bm0   = blockIdx.y * BM;
    const int bn0   = blockIdx.x * BN;

    f32x4 areg[8];
    auto loadA = [&](int k0) {
#pragma unroll
        for (int u = 0; u < 8; ++u) {
            int c = tid + u * 256, r = c >> 4, col = (c & 15) * 4;
            areg[u] = *(const f32x4*)(&A[(size_t)(bm0 + r) * K + k0 + col]);
        }
    };
    auto storeA = [&]() {
#pragma unroll
        for (int u = 0; u < 8; ++u) {
            int c = tid + u * 256, r = c >> 4, col = (c & 15) * 4;
            bf16x4 o;
            o[0] = (__bf16)areg[u].x; o[1] = (__bf16)areg[u].y;
            o[2] = (__bf16)areg[u].z; o[3] = (__bf16)areg[u].w;
            *(bf16x4*)(&As[r * LDT + col]) = o;
        }
    };
    auto loadB = [&](int buf, int k0) {
#pragma unroll
        for (int c = tid; c < BN * 8; c += 256) {
            int r = c >> 3, col = (c & 7) * 8;
            async_copy_b128(&B[(size_t)(bn0 + r) * K + k0 + col],
                            &Bs[buf][r * LDT + col]);
        }
    };

    f32x8 acc[4][2];
    const f32x8 zacc = {0.f, 0.f, 0.f, 0.f, 0.f, 0.f, 0.f, 0.f};
#pragma unroll
    for (int i = 0; i < 4; ++i)
#pragma unroll
        for (int j = 0; j < 2; ++j) acc[i][j] = zacc;

    // Prologue: fill As and Bs[0]
    loadA(0);
    loadB(0, 0);
    storeA();
    async_wait0();
    __syncthreads();

    int buf = 0;
    for (int k0 = 0; k0 < K; k0 += BK) {
        const int kn = k0 + BK;
        if (kn < K) {
            loadA(kn);          // next A tile -> registers (overlaps compute)
            loadB(buf ^ 1, kn); // next B tile -> LDS via async DMA
        }

#pragma unroll
        for (int ks = 0; ks < BK; ks += 32) {
            ABf af[4], bfr[2];
#pragma unroll
            for (int i = 0; i < 4; ++i) {
                // A frag 16x32: lanes 0-15 K 0..15; lanes 16-31 K 16..31
                int r = wm * 64 + i * 16 + l16;
                af[i].h[0] = *(const bf16x8*)(&As[r * LDT + ks + lhalf * 16]);
                af[i].h[1] = *(const bf16x8*)(&As[r * LDT + ks + lhalf * 16 + 8]);
            }
#pragma unroll
            for (int j = 0; j < 2; ++j) {
                // B frag 32x16: lanes 0-15 K 0..15 at N=lane; lanes 16-31 K 16..31
                int r = wn * 32 + j * 16 + l16;
                bfr[j].h[0] = *(const bf16x8*)(&Bs[buf][r * LDT + ks + lhalf * 16]);
                bfr[j].h[1] = *(const bf16x8*)(&Bs[buf][r * LDT + ks + lhalf * 16 + 8]);
            }
#pragma unroll
            for (int i = 0; i < 4; ++i)
#pragma unroll
                for (int j = 0; j < 2; ++j)
                    acc[i][j] = wmma_bf16(af[i].v, bfr[j].v, acc[i][j]);
        }

        if (kn < K) {
            __syncthreads();    // all waves done reading As for this k-step
            storeA();           // overwrite As with next tile
            async_wait0();      // next Bs buffer landed in LDS
        }
        __syncthreads();
        buf ^= 1;
    }

    // Epilogue. D layout: VGPR r -> (M=r, N=l16) lanes 0-15 ; (M=r+8, N=l16) lanes 16-31
#pragma unroll
    for (int i = 0; i < 4; ++i)
#pragma unroll
        for (int j = 0; j < 2; ++j)
#pragma unroll
            for (int r = 0; r < 8; ++r) {
                int gm = bm0 + wm * 64 + i * 16 + r + lhalf * 8;
                int gn = bn0 + wn * 32 + j * 16 + l16;
                float val = acc[i][j][r];
                if (MODE == 0) {
                    int b_ = gm / SEQ, s_ = gm % SEQ;
                    int h_ = gn / DK,  d_ = gn % DK;
                    ((__bf16*)out)[(((size_t)b_ * NHEAD + h_) * SEQ + s_) * DK + d_] =
                        (__bf16)val;
                } else {
                    ((float*)out)[(size_t)gm * N + gn] = val;
                }
            }
}

// ---------------------------------------------------------------------------
// Flash attention per (b,h), 64-query block, 4 wave32 waves (16 q-rows each).
// Q,K,V bf16 in [B,H,S,DK]. Online softmax with masking. Output f32 merged
// heads: aout[b*S+s, h*DK+d].
// ---------------------------------------------------------------------------
__global__ __launch_bounds__(128) void flash_attn_wmma(const __bf16* __restrict__ Qh,
                                                       const __bf16* __restrict__ Kh,
                                                       const __bf16* __restrict__ Vh,
                                                       const int* __restrict__ mask,
                                                       float* __restrict__ aout) {
    constexpr int LKT = DK + 8;
    __shared__ __bf16 Ks[64 * LKT];          // K tile, row-major [key][d]
    __shared__ __bf16 Vt[64 * LKT];          // V tile, transposed [d][key]
    __shared__ __bf16 Ps[4 * 16 * LKT];      // per-wave P tile [16 rows][64 keys]

    const int tid   = threadIdx.x;
    const int wave  = tid >> 5;
    const int lane  = tid & 31;
    const int l16   = lane & 15;
    const int lhalf = lane >> 4;

    const int bh = blockIdx.y;
    const int b  = bh >> 4;           // NHEAD == 16
    const int h  = bh & 15;
    const int q0 = blockIdx.x * 64;
    const size_t base = (size_t)bh * SEQ * DK;

    // Q fragments: 16 rows per wave, two 32-wide K(=d) steps, resident in VGPRs
    ABf qf[2];
    {
        const __bf16* qp = Qh + base + (size_t)(q0 + wave * 16 + l16) * DK;
#pragma unroll
        for (int st = 0; st < 2; ++st) {
            qf[st].h[0] = *(const bf16x8*)(qp + st * 32 + lhalf * 16);
            qf[st].h[1] = *(const bf16x8*)(qp + st * 32 + lhalf * 16 + 8);
        }
    }

    f32x8 accO[4];
    const f32x8 zacc = {0.f, 0.f, 0.f, 0.f, 0.f, 0.f, 0.f, 0.f};
#pragma unroll
    for (int t = 0; t < 4; ++t) accO[t] = zacc;
    float mrow[8], lrow[8];
#pragma unroll
    for (int r = 0; r < 8; ++r) { mrow[r] = -1e30f; lrow[r] = 0.f; }

    for (int kb = 0; kb < SEQ; kb += 64) {
        // Stage K tile row-major via async DMA (bf16, no conversion needed)
#pragma unroll
        for (int c = tid; c < 512; c += 128) {
            int r = c >> 3, col = (c & 7) * 8;
            async_copy_b128(Kh + base + (size_t)(kb + r) * DK + col,
                            &Ks[r * LKT + col]);
        }
        // Stage V tile transposed: Vt[d][key] (through VGPRs)
#pragma unroll
        for (int c = tid; c < 512; c += 128) {
            int r = c >> 3, col = (c & 7) * 8;
            bf16x8 v = *(const bf16x8*)(Vh + base + (size_t)(kb + r) * DK + col);
#pragma unroll
            for (int e = 0; e < 8; ++e) Vt[(col + e) * LKT + r] = v[e];
        }
        async_wait0();
        __syncthreads();

        // scores = Q * K^T : 4 n-tiles of 16 keys, 2 k-steps over d
        f32x8 sc[4];
#pragma unroll
        for (int t = 0; t < 4; ++t) sc[t] = zacc;
#pragma unroll
        for (int t = 0; t < 4; ++t)
#pragma unroll
            for (int st = 0; st < 2; ++st) {
                ABf bk;
                int krow = t * 16 + l16;   // key index within tile == output N
                bk.h[0] = *(const bf16x8*)(&Ks[krow * LKT + st * 32 + lhalf * 16]);
                bk.h[1] = *(const bf16x8*)(&Ks[krow * LKT + st * 32 + lhalf * 16 + 8]);
                sc[t] = wmma_bf16(qf[st].v, bk.v, sc[t]);
            }

        // scale + key mask (column = kb + t*16 + l16 for both lane halves)
#pragma unroll
        for (int t = 0; t < 4; ++t) {
            bool dead = (mask[b * SEQ + kb + t * 16 + l16] == 0);
#pragma unroll
            for (int r = 0; r < 8; ++r)
                sc[t][r] = dead ? -1e9f : sc[t][r] * 0.125f;   // 1/sqrt(64)
        }

        // online softmax: row stats via xor-shuffle inside each 16-lane half
        float mnew[8], alpha[8];
#pragma unroll
        for (int r = 0; r < 8; ++r) {
            float mx = fmaxf(fmaxf(sc[0][r], sc[1][r]), fmaxf(sc[2][r], sc[3][r]));
#pragma unroll
            for (int off = 1; off < 16; off <<= 1)
                mx = fmaxf(mx, __shfl_xor(mx, off, 32));
            mnew[r]  = fmaxf(mrow[r], mx);
            alpha[r] = __expf(mrow[r] - mnew[r]);
            mrow[r]  = mnew[r];
        }
#pragma unroll
        for (int r = 0; r < 8; ++r) {
            float sum = 0.f;
#pragma unroll
            for (int t = 0; t < 4; ++t) {
                float p = __expf(sc[t][r] - mnew[r]);
                sc[t][r] = p;
                sum += p;
            }
#pragma unroll
            for (int off = 1; off < 16; off <<= 1)
                sum += __shfl_xor(sum, off, 32);
            lrow[r] = alpha[r] * lrow[r] + sum;
#pragma unroll
            for (int t = 0; t < 4; ++t) accO[t][r] *= alpha[r];
        }

        // P (D layout) -> LDS bf16 so it can be reloaded in A-fragment layout
        __bf16* pw = &Ps[wave * 16 * LKT];
#pragma unroll
        for (int t = 0; t < 4; ++t)
#pragma unroll
            for (int r = 0; r < 8; ++r)
                pw[(r + lhalf * 8) * LKT + t * 16 + l16] = (__bf16)sc[t][r];
        __syncthreads();

        // O += P * V : contraction over 64 keys (2 k-steps), 4 d-tiles
#pragma unroll
        for (int st = 0; st < 2; ++st) {
            ABf pa;
            pa.h[0] = *(const bf16x8*)(&pw[l16 * LKT + st * 32 + lhalf * 16]);
            pa.h[1] = *(const bf16x8*)(&pw[l16 * LKT + st * 32 + lhalf * 16 + 8]);
#pragma unroll
            for (int td = 0; td < 4; ++td) {
                ABf bv;
                int drow = td * 16 + l16;   // output d == N
                bv.h[0] = *(const bf16x8*)(&Vt[drow * LKT + st * 32 + lhalf * 16]);
                bv.h[1] = *(const bf16x8*)(&Vt[drow * LKT + st * 32 + lhalf * 16 + 8]);
                accO[td] = wmma_bf16(pa.v, bv.v, accO[td]);
            }
        }
        __syncthreads();
    }

    // epilogue: normalize and write merged-head f32 activations
#pragma unroll
    for (int r = 0; r < 8; ++r) {
        float inv = 1.0f / lrow[r];
        int srow = q0 + wave * 16 + r + lhalf * 8;
        size_t orow = ((size_t)b * SEQ + srow) * DMODEL + h * DK;
#pragma unroll
        for (int td = 0; td < 4; ++td)
            aout[orow + td * 16 + l16] = accO[td][r] * inv;
    }
}

// ---------------------------------------------------------------------------
// Host launcher
// ---------------------------------------------------------------------------
extern "C" void kernel_launch(void* const* d_in, const int* in_sizes, int n_in,
                              void* d_out, int out_size, void* d_ws, size_t ws_size,
                              hipStream_t stream) {
    const float* q    = (const float*)d_in[0];
    const float* k    = (const float*)d_in[1];
    const float* v    = (const float*)d_in[2];
    const int*   mask = (const int*)d_in[3];
    const float* Wq   = (const float*)d_in[4];
    const float* Wk   = (const float*)d_in[5];
    const float* Wv   = (const float*)d_in[6];
    const float* Wo   = (const float*)d_in[7];

    char* ws = (char*)d_ws;
    const size_t MB = 1ull << 20;
    __bf16* wqb  = (__bf16*)(ws + 0 * MB);   // 2 MB each
    __bf16* wkb  = (__bf16*)(ws + 2 * MB);
    __bf16* wvb  = (__bf16*)(ws + 4 * MB);
    __bf16* wob  = (__bf16*)(ws + 6 * MB);
    __bf16* Qh   = (__bf16*)(ws + 8 * MB);   // 16 MB each, [B,H,S,DK] bf16
    __bf16* Kh   = (__bf16*)(ws + 24 * MB);
    __bf16* Vh   = (__bf16*)(ws + 40 * MB);
    float*  aout = (float*)(ws + 56 * MB);   // 32 MB, [B*S, DMODEL] f32

    const int NW = DMODEL * DMODEL;          // 1M weight elements
    cast_f32_bf16<<<NW / 1024, 256, 0, stream>>>(Wq, wqb, NW);
    cast_f32_bf16<<<NW / 1024, 256, 0, stream>>>(Wk, wkb, NW);
    cast_f32_bf16<<<NW / 1024, 256, 0, stream>>>(Wv, wvb, NW);
    cast_f32_bf16<<<NW / 1024, 256, 0, stream>>>(Wo, wob, NW);

    const int M = BATCH * SEQ;               // 8192
    dim3 ggrid(DMODEL / 128, M / 128);       // (8, 64)
    gemm_bf16_wmma<0><<<ggrid, 256, 0, stream>>>(q, wqb, Qh, M, DMODEL, DMODEL);
    gemm_bf16_wmma<0><<<ggrid, 256, 0, stream>>>(k, wkb, Kh, M, DMODEL, DMODEL);
    gemm_bf16_wmma<0><<<ggrid, 256, 0, stream>>>(v, wvb, Vh, M, DMODEL, DMODEL);

    dim3 agrid(SEQ / 64, BATCH * NHEAD);     // (32, 64)
    flash_attn_wmma<<<agrid, 128, 0, stream>>>(Qh, Kh, Vh, mask, aout);

    gemm_bf16_wmma<1><<<ggrid, 256, 0, stream>>>(aout, wob, d_out, M, DMODEL, DMODEL);
}
